// HGNN_33045478376091
// MI455X (gfx1250) — compile-verified, matrix operand order checked by reading, not run on hardware
//
#include <hip/hip_runtime.h>
#include <hip/hip_bf16.h>
#include <stdint.h>

typedef __attribute__((ext_vector_type(16))) __bf16 v16bf;
typedef __attribute__((ext_vector_type(8)))  float  v8f;

#define D 128
#define SLOPE 0.2f

__device__ __forceinline__ unsigned short f2bf(float f) {
    union { float f; unsigned u; } v; v.f = f;
    unsigned u = v.u;
    u += 0x7FFFu + ((u >> 16) & 1u);     // round-to-nearest-even
    return (unsigned short)(u >> 16);
}

union FragAB { uint4 q[2]; v16bf v; };

// ---------------------------------------------------------------------------
// Pre-swizzle a 128x128 fp32 weight (row-major, K x N) into bf16 B-matrix
// fragments matching v_wmma_f32_16x16x32_bf16 SRC1 lane layout:
//   lane l: K = kb..kb+7 in elems 0..7, K = kb+16..kb+23 in elems 8..15
//   with kb = 32*ktile + 8*(l>>4); column N = 16*ntile + (l&15).
// One block of 32 lanes per (matrix, ktile, ntile).
// ---------------------------------------------------------------------------
__global__ void k_prep_w(const float* __restrict__ w0, const float* __restrict__ w1,
                         const float* __restrict__ w2, const float* __restrict__ w3,
                         const float* __restrict__ w4, const float* __restrict__ w5,
                         unsigned short* __restrict__ wf) {
    int b   = blockIdx.x;           // [0, 6*32)
    int mat = b >> 5;
    int rem = b & 31;
    int t   = rem >> 3;             // ktile 0..3
    int n   = rem & 7;              // ntile 0..7
    const float* w = (mat == 0) ? w0 : (mat == 1) ? w1 : (mat == 2) ? w2
                   : (mat == 3) ? w3 : (mat == 4) ? w4 : w5;
    int l = threadIdx.x;            // lane 0..31
    unsigned short* dst = wf + (size_t)mat * 16384 + (((t * 8 + n) * 32 + l) * 16);
    int kb  = 32 * t + 8 * (l >> 4);
    int col = 16 * n + (l & 15);
    #pragma unroll
    for (int i = 0; i < 16; ++i) {
        int k = kb + (i & 7) + ((i >> 3) << 4);
        dst[i] = f2bf(w[k * D + col]);
    }
}

// ---------------------------------------------------------------------------
// Projection GEMM: out[M,128] = x[M,128] @ W[128,128] + bias, bf16 WMMA.
// 128 threads = 4 waves; block covers 64 rows; wave covers 16 rows x 128 cols.
// Full-tile fast path (781/782 blocks): no per-element bounds checks.
// ---------------------------------------------------------------------------
__global__ __launch_bounds__(128)
void k_proj(const float* __restrict__ x, const unsigned short* __restrict__ wf,
            const float* __restrict__ bias, float* __restrict__ out, int M) {
    __shared__ __align__(16) unsigned short lds[64][136];   // +8 pad: kill bank conflicts
    int r0   = blockIdx.x * 64;
    int tid  = threadIdx.x;
    bool full = (r0 + 64 <= M);                             // block-uniform predicate

    // Stage 64x128 fp32 -> bf16 into LDS (coalesced b128 global loads).
    if (full) {
        #pragma unroll 4
        for (int it = 0; it < 16; ++it) {
            int idx = tid + 128 * it;        // float4 chunk id, 0..2047
            int row = idx >> 5;              // 32 float4 per row
            int cc  = (idx & 31) * 4;
            float4 v = ((const float4*)(x + (size_t)(r0 + row) * D))[idx & 31];
            unsigned lo = (unsigned)f2bf(v.x) | ((unsigned)f2bf(v.y) << 16);
            unsigned hi = (unsigned)f2bf(v.z) | ((unsigned)f2bf(v.w) << 16);
            *(uint2*)&lds[row][cc] = make_uint2(lo, hi);
        }
    } else {
        #pragma unroll 4
        for (int it = 0; it < 16; ++it) {
            int idx = tid + 128 * it;
            int row = idx >> 5;
            int cc  = (idx & 31) * 4;
            int g   = r0 + row;
            float4 v = make_float4(0.f, 0.f, 0.f, 0.f);
            if (g < M) v = ((const float4*)(x + (size_t)g * D))[idx & 31];
            unsigned lo = (unsigned)f2bf(v.x) | ((unsigned)f2bf(v.y) << 16);
            unsigned hi = (unsigned)f2bf(v.z) | ((unsigned)f2bf(v.w) << 16);
            *(uint2*)&lds[row][cc] = make_uint2(lo, hi);
        }
    }
    __syncthreads();

    int w    = tid >> 5;                 // wave 0..3
    int l    = tid & 31;                 // lane
    int mrow = 16 * w + (l & 15);        // A-matrix row for this lane
    int hsel = l >> 4;                   // K-half select

    // A fragments for all 4 ktiles (ISA 16-bit A 16x32 layout).
    FragAB a[4];
    #pragma unroll
    for (int t = 0; t < 4; ++t) {
        int k0 = 32 * t + 8 * hsel;
        a[t].q[0] = *(const uint4*)&lds[mrow][k0];
        a[t].q[1] = *(const uint4*)&lds[mrow][k0 + 16];
    }

    const uint4* wq = (const uint4*)wf;  // 2 uint4 per (ktile,ntile,lane)
    #pragma unroll
    for (int n = 0; n < 8; ++n) {
        v8f c = {};
        #pragma unroll
        for (int t = 0; t < 4; ++t) {
            FragAB b;
            size_t fi = ((size_t)(t * 8 + n) * 32 + l) * 2;
            b.q[0] = wq[fi];
            b.q[1] = wq[fi + 1];
            c = __builtin_amdgcn_wmma_f32_16x16x32_bf16(false, a[t].v, false, b.v,
                                                        (short)0, c, false, false);
        }
        int   col = 16 * n + (l & 15);
        float bv  = bias[col];
        float* ob = out + (size_t)(r0 + 16 * w + 8 * hsel) * D + col;
        if (full) {
            #pragma unroll
            for (int r = 0; r < 8; ++r)      // straight clause of global_store_b32
                ob[(size_t)r * D] = c[r] + bv;
        } else {
            #pragma unroll
            for (int r = 0; r < 8; ++r) {    // C/D layout: lanes16-31 hold M = r+8
                int row = r0 + 16 * w + r + 8 * hsel;
                if (row < M) out[(size_t)row * D + col] = c[r] + bv;
            }
        }
    }
}

// ---------------------------------------------------------------------------
// Per-edge attention logit + running segment max.  One wave per edge.
// e[k] = att . LeakyReLU(hl[src] + hr[dst])
// ---------------------------------------------------------------------------
__global__ void k_att(const float* __restrict__ hl, const float* __restrict__ hr,
                      const int* __restrict__ src, const int* __restrict__ dst,
                      const float* __restrict__ att,
                      float* __restrict__ e, float* __restrict__ m, int E) {
    int wid = (blockIdx.x * blockDim.x + threadIdx.x) >> 5;
    int l   = threadIdx.x & 31;
    if (wid >= E) return;
    int s = src[wid], d = dst[wid];
    float4 a4  = ((const float4*)att)[l];
    float4 hl4 = ((const float4*)(hl + (size_t)s * D))[l];
    float4 hr4 = ((const float4*)(hr + (size_t)d * D))[l];
    float p = 0.f, t;
    t = hl4.x + hr4.x; p += a4.x * (t > 0.f ? t : SLOPE * t);
    t = hl4.y + hr4.y; p += a4.y * (t > 0.f ? t : SLOPE * t);
    t = hl4.z + hr4.z; p += a4.z * (t > 0.f ? t : SLOPE * t);
    t = hl4.w + hr4.w; p += a4.w * (t > 0.f ? t : SLOPE * t);
    #pragma unroll
    for (int o = 16; o > 0; o >>= 1) p += __shfl_xor(p, o, 32);
    if (l == 0) {
        e[wid] = p;
        // float-max via signed/unsigned integer atomics (works with -inf init)
        if (p >= 0.f) atomicMax((int*)(m + d), __float_as_int(p));
        else          atomicMin((unsigned int*)(m + d), __float_as_uint(p));
    }
}

// ---------------------------------------------------------------------------
// a = exp(e - m[dst]); overwrite e with a; accumulate softmax denominator.
// ---------------------------------------------------------------------------
__global__ void k_exp(const int* __restrict__ dst, float* __restrict__ e,
                      const float* __restrict__ m, float* __restrict__ den, int E) {
    int k = blockIdx.x * blockDim.x + threadIdx.x;
    if (k >= E) return;
    int d = dst[k];
    float a = __expf(e[k] - m[d]);
    e[k] = a;
    atomicAdd(den + d, a);
}

// ---------------------------------------------------------------------------
// Weighted scatter: acc[dst] += (a/denom[dst]) * hl[src].  One wave per edge,
// 4 global f32 atomics per lane (L2-side atomic units, L2-resident table).
// ---------------------------------------------------------------------------
__global__ void k_scatter(const float* __restrict__ hl, const int* __restrict__ src,
                          const int* __restrict__ dst, const float* __restrict__ e,
                          const float* __restrict__ den, float* __restrict__ acc, int E) {
    int wid = (blockIdx.x * blockDim.x + threadIdx.x) >> 5;
    int l   = threadIdx.x & 31;
    if (wid >= E) return;
    int s = src[wid], d = dst[wid];
    float alpha = e[wid] / den[d];
    float4 h4 = ((const float4*)(hl + (size_t)s * D))[l];
    float* o = acc + (size_t)d * D + 4 * l;
    atomicAdd(o + 0, alpha * h4.x);
    atomicAdd(o + 1, alpha * h4.y);
    atomicAdd(o + 2, alpha * h4.z);
    atomicAdd(o + 3, alpha * h4.w);
}

// ---------------------------------------------------------------------------
// Workspace (re)initialization each call: accumulators=0, maxes=-inf, den=0.
// ---------------------------------------------------------------------------
__global__ void k_init(float* __restrict__ acc_a, float* __restrict__ acc_b,
                       float* __restrict__ m3, float* __restrict__ den3,
                       int na, int nb, int nm) {
    int i = blockIdx.x * blockDim.x + threadIdx.x;
    int stride = gridDim.x * blockDim.x;
    for (int k = i; k < na; k += stride) acc_a[k] = 0.f;
    for (int k = i; k < nb; k += stride) acc_b[k] = 0.f;
    for (int k = i; k < nm; k += stride) { ((int*)m3)[k] = (int)0xFF800000; den3[k] = 0.f; }
}

// ---------------------------------------------------------------------------
// HeteroConv mean over relations + per-relation bias + final ReLU.
// d_out = [ out_a (N_A*128) | out_b (N_B*128) ]
// ---------------------------------------------------------------------------
__global__ void k_final(const float* __restrict__ acc_a, const float* __restrict__ acc_b,
                        const float* __restrict__ bias_ba, const float* __restrict__ bias_aa,
                        const float* __restrict__ bias_ab,
                        float* __restrict__ out, int na, int nb) {
    int i = blockIdx.x * blockDim.x + threadIdx.x;
    if (i < na) {
        int c = i & (D - 1);
        float v = 0.5f * (acc_a[i] + bias_ba[c] + bias_aa[c]);
        out[i] = v > 0.f ? v : 0.f;
    } else if (i < na + nb) {
        int j = i - na;
        int c = j & (D - 1);
        float v = acc_b[j] + bias_ab[c];
        out[na + j] = v > 0.f ? v : 0.f;
    }
}

extern "C" void kernel_launch(void* const* d_in, const int* in_sizes, int n_in,
                              void* d_out, int out_size, void* d_ws, size_t ws_size,
                              hipStream_t stream) {
    (void)n_in; (void)out_size; (void)ws_size;
    const float* x_a = (const float*)d_in[0];
    const float* x_b = (const float*)d_in[1];
    const int* edge_ab = (const int*)d_in[2];
    const int* edge_ba = (const int*)d_in[3];
    const int* edge_aa = (const int*)d_in[4];
    const float *Wl_ab = (const float*)d_in[5],  *bl_ab = (const float*)d_in[6],
                *Wr_ab = (const float*)d_in[7],  *br_ab = (const float*)d_in[8],
                *att_ab = (const float*)d_in[9], *bias_ab = (const float*)d_in[10];
    const float *Wl_ba = (const float*)d_in[11], *bl_ba = (const float*)d_in[12],
                *Wr_ba = (const float*)d_in[13], *br_ba = (const float*)d_in[14],
                *att_ba = (const float*)d_in[15], *bias_ba = (const float*)d_in[16];
    const float *Wl_aa = (const float*)d_in[17], *bl_aa = (const float*)d_in[18],
                *Wr_aa = (const float*)d_in[19], *br_aa = (const float*)d_in[20],
                *att_aa = (const float*)d_in[21], *bias_aa = (const float*)d_in[22];

    const int MA = in_sizes[0] / D;
    const int MB = in_sizes[1] / D;
    const int E  = in_sizes[2] / 2;

    // ---- workspace layout (floats) ----
    float* f = (float*)d_ws;
    size_t off = 0;
    float* hl_ab = f + off; off += (size_t)MA * D;   // src a
    float* hr_ab = f + off; off += (size_t)MB * D;   // dst b
    float* hl_ba = f + off; off += (size_t)MB * D;   // src b
    float* hr_ba = f + off; off += (size_t)MA * D;   // dst a
    float* hl_aa = f + off; off += (size_t)MA * D;   // src a
    float* hr_aa = f + off; off += (size_t)MA * D;   // dst a
    float* acc_a = f + off; off += (size_t)MA * D;
    float* acc_b = f + off; off += (size_t)MB * D;
    float* eab = f + off; off += (size_t)E;
    float* eba = f + off; off += (size_t)E;
    float* eaa = f + off; off += (size_t)E;
    float* m3  = f + off; off += (size_t)(MB + 2 * MA);
    float* d3  = f + off; off += (size_t)(MB + 2 * MA);
    float* m_ab = m3;           float* m_ba = m3 + MB;  float* m_aa = m3 + MB + MA;
    float* d_ab = d3;           float* d_ba = d3 + MB;  float* d_aa = d3 + MB + MA;
    off = (off + 3) & ~(size_t)3;                     // 16B align for uint4 loads
    unsigned short* wf = (unsigned short*)(f + off);  // 6 * 16384 bf16 fragments

    const int* s_ab = edge_ab; const int* t_ab = edge_ab + E;
    const int* s_ba = edge_ba; const int* t_ba = edge_ba + E;
    const int* s_aa = edge_aa; const int* t_aa = edge_aa + E;

    // 1) init accumulators / maxes / denominators
    k_init<<<2048, 256, 0, stream>>>(acc_a, acc_b, m3, d3, MA * D, MB * D, MB + 2 * MA);

    // 2) pre-swizzle weights to bf16 WMMA fragments
    k_prep_w<<<6 * 32, 32, 0, stream>>>(Wl_ab, Wr_ab, Wl_ba, Wr_ba, Wl_aa, Wr_aa, wf);

    // 3) six projections (bf16 WMMA GEMMs)
    int ga = (MA + 63) / 64, gb = (MB + 63) / 64;
    k_proj<<<ga, 128, 0, stream>>>(x_a, wf + 0 * 16384, bl_ab, hl_ab, MA);
    k_proj<<<gb, 128, 0, stream>>>(x_b, wf + 1 * 16384, br_ab, hr_ab, MB);
    k_proj<<<gb, 128, 0, stream>>>(x_b, wf + 2 * 16384, bl_ba, hl_ba, MB);
    k_proj<<<ga, 128, 0, stream>>>(x_a, wf + 3 * 16384, br_ba, hr_ba, MA);
    k_proj<<<ga, 128, 0, stream>>>(x_a, wf + 4 * 16384, bl_aa, hl_aa, MA);
    k_proj<<<ga, 128, 0, stream>>>(x_a, wf + 5 * 16384, br_aa, hr_aa, MA);

    // 4) attention logits + segment max (wave per edge)
    int gw = (int)(((long long)E * 32 + 255) / 256);
    k_att<<<gw, 256, 0, stream>>>(hl_ab, hr_ab, s_ab, t_ab, att_ab, eab, m_ab, E);
    k_att<<<gw, 256, 0, stream>>>(hl_ba, hr_ba, s_ba, t_ba, att_ba, eba, m_ba, E);
    k_att<<<gw, 256, 0, stream>>>(hl_aa, hr_aa, s_aa, t_aa, att_aa, eaa, m_aa, E);

    // 5) exp + softmax denominators
    int ge = (E + 255) / 256;
    k_exp<<<ge, 256, 0, stream>>>(t_ab, eab, m_ab, d_ab, E);
    k_exp<<<ge, 256, 0, stream>>>(t_ba, eba, m_ba, d_ba, E);
    k_exp<<<ge, 256, 0, stream>>>(t_aa, eaa, m_aa, d_aa, E);

    // 6) weighted scatter-sum (both a-destined relations share acc_a)
    k_scatter<<<gw, 256, 0, stream>>>(hl_ab, s_ab, t_ab, eab, d_ab, acc_b, E);
    k_scatter<<<gw, 256, 0, stream>>>(hl_ba, s_ba, t_ba, eba, d_ba, acc_a, E);
    k_scatter<<<gw, 256, 0, stream>>>(hl_aa, s_aa, t_aa, eaa, d_aa, acc_a, E);

    // 7) mean over relations + biases + ReLU -> d_out
    int tot = (MA + MB) * D;
    k_final<<<(tot + 255) / 256, 256, 0, stream>>>(acc_a, acc_b, bias_ba, bias_aa, bias_ab,
                                                   (float*)d_out, MA * D, MB * D);
}
